// MaskedMultiheadAttention_38680475467867
// MI455X (gfx1250) — compile-verified, matrix operand order checked by reading, not run on hardware
//
#include <hip/hip_runtime.h>
#include <hip/hip_bf16.h>

typedef __attribute__((ext_vector_type(16))) __bf16 v16bf;
typedef __attribute__((ext_vector_type(8)))  float  v8f;
typedef __attribute__((ext_vector_type(4)))  unsigned int u32x4;
typedef __attribute__((ext_vector_type(8)))  int i32x8;
typedef __attribute__((ext_vector_type(4)))  int i32x4;

#define S_LEN 2048
#define D_DIM 1024
#define H_NUM 16
#define HS    64
#define B_NUM 2

#if defined(__has_builtin)
#if __has_builtin(__builtin_amdgcn_tensor_load_to_lds)
#define HAVE_TDM 1
#endif
#endif

#if __has_include(<hip/amd_detail/amd_gfx1250_TDM.h>)
#define TDM_6ARG 1
#endif

__device__ __forceinline__ v8f wmma_bf16(v16bf a, v16bf b, v8f c) {
  return __builtin_amdgcn_wmma_f32_16x16x32_bf16(false, a, false, b, (short)0, c,
                                                 false, false);
}

#if defined(HAVE_TDM)
// Issue a TDM 2-D tile load (bf16 elements) into LDS. Wave-level op; EXEC ignored.
// D# per CDNA5 ISA: group0 = count/lds/global/type,
// group1 = data_size + tensor/tile dims + dim0 stride. Groups 2/3 zero (<=2D).
__device__ __forceinline__ void tdm_load_tile_2d(unsigned lds_byte_addr,
                                                 const void* global_tile,
                                                 unsigned tile_w,      // dim0, elems
                                                 unsigned tile_h,      // dim1, rows
                                                 unsigned long long row_stride) {  // elems
  unsigned long long ga = (unsigned long long)global_tile;
  u32x4 g0 = {0u, 0u, 0u, 0u};
  g0.x = 1u;                                   // count=1 (valid), user mode
  g0.y = lds_byte_addr;                        // lds_addr
  g0.z = (unsigned)ga;                         // global_addr[31:0]
  g0.w = (unsigned)((ga >> 32) & 0x1ffffffu) | (2u << 30);  // addr[56:32], type=2
  i32x8 g1 = {0, 0, 0, 0, 0, 0, 0, 0};
  g1[0] = (int)(1u << 16);                     // data_size=1 -> 2 bytes, mask=0
  g1[1] = (int)((tile_w & 0xffffu) << 16);     // tensor_dim0[15:0] (== tile_w)
  g1[2] = (int)(((tile_w >> 16) & 0xffffu) | ((tile_h & 0xffffu) << 16));  // dim0 hi, dim1 lo
  g1[3] = (int)(((tile_h >> 16) & 0xffffu) | ((tile_w & 0xffffu) << 16));  // dim1 hi, tile_dim0
  g1[4] = (int)(tile_h & 0xffffu);             // tile_dim1 (tile_dim2=0)
  g1[5] = (int)(unsigned)(row_stride & 0xffffffffull);        // dim0_stride[31:0]
  g1[6] = (int)(unsigned)((row_stride >> 32) & 0xffffull);    // dim0_stride[47:32]
  i32x4 z4 = {0, 0, 0, 0};
#if defined(TDM_6ARG)
  i32x8 z8 = {0, 0, 0, 0, 0, 0, 0, 0};
  __builtin_amdgcn_tensor_load_to_lds(g0, g1, z4, z4, z8, 0);
#else
  __builtin_amdgcn_tensor_load_to_lds(g0, g1, z4, z4, 0);
#endif
}
#endif  // HAVE_TDM

// ---------------- f32 -> bf16 converts ----------------
__global__ __launch_bounds__(256) void cvt_bf16_kernel(const float* __restrict__ in,
                                                       __bf16* __restrict__ out, int n) {
  int i = blockIdx.x * 256 + threadIdx.x;
  if (i < n) out[i] = (__bf16)in[i];
}

// (H,R,C) f32 -> (H,C,R) bf16 (transpose last two dims)
__global__ __launch_bounds__(256) void cvt_bf16_tr_kernel(const float* __restrict__ in,
                                                          __bf16* __restrict__ out,
                                                          int H, int R, int C) {
  int i = blockIdx.x * 256 + threadIdx.x;
  int total = H * R * C;
  if (i >= total) return;
  int h = i / (R * C);
  int rc = i - h * R * C;
  int r = rc / C;
  int c = rc - r * C;
  out[(size_t)h * C * R + (size_t)c * R + r] = (__bf16)in[i];
}

// ---------------- QKV projection ----------------
// Q[b,h] (S x hs) = X[b] (S x D) @ Wq[h] (D x hs); same for K, V.
// Q,K row-major (S x hs); V transposed (hs x S). 1/sqrt(hs) folded into Q.
// X tile staged by the Tensor Data Mover; weight tiles by vector loads.
__global__ __launch_bounds__(256) void qkv_kernel(const __bf16* __restrict__ xb,
                                                  const __bf16* __restrict__ wqT,
                                                  const __bf16* __restrict__ wkT,
                                                  const __bf16* __restrict__ wvT,
                                                  __bf16* __restrict__ qb,
                                                  __bf16* __restrict__ kb,
                                                  __bf16* __restrict__ vTb) {
  const int LDX = 32;  // X tile row stride (TDM packs rows contiguously)
  const int LDW = 48;  // weight tile row stride (padded)
  __shared__ __attribute__((aligned(32))) __bf16 xs[128 * LDX];
  __shared__ __attribute__((aligned(32))) __bf16 wls[3][64 * LDW];

  int bh = blockIdx.x;
  int b = bh >> 4, h = bh & 15;
  int s0 = blockIdx.y * 128;
  int tid = threadIdx.x;
  int lane = tid & 31, wave = tid >> 5;

  const __bf16* xrow = xb + (size_t)b * S_LEN * D_DIM;
  const __bf16* wT[3] = {wqT, wkT, wvT};

  v8f accq[4], acck[4], accv[4];
#pragma unroll
  for (int i = 0; i < 4; ++i) { accq[i] = (v8f)(0.f); acck[i] = (v8f)(0.f); accv[i] = (v8f)(0.f); }

  for (int k0 = 0; k0 < D_DIM; k0 += 32) {
    // X tile: 128 rows x 32 elems via TDM (one wave issues, DMA fills LDS)
#if defined(HAVE_TDM)
    if (wave == 0) {
      tdm_load_tile_2d((unsigned)(size_t)&xs[0],
                       &xrow[(size_t)s0 * D_DIM + k0], 32u, 128u,
                       (unsigned long long)D_DIM);
      __builtin_amdgcn_s_wait_tensorcnt(0);
    }
#else
#pragma unroll
    for (int it = 0; it < 2; ++it) {
      int i = tid + it * 256;
      int r = i >> 2, q = i & 3;
      *(float4*)&xs[r * LDX + q * 8] =
          *(const float4*)&xrow[(size_t)(s0 + r) * D_DIM + k0 + q * 8];
    }
#endif
    // weight tiles: 64 x 32 each, one 16B chunk per thread per tensor
    {
      int n = tid >> 2, q = tid & 3;
#pragma unroll
      for (int t = 0; t < 3; ++t)
        *(float4*)&wls[t][n * LDW + q * 8] =
            *(const float4*)&wT[t][(size_t)(h * HS + n) * D_DIM + k0 + q * 8];
    }
    __syncthreads();

    int koff = (lane >> 4) * 16;
    v16bf a = *(const v16bf*)&xs[(wave * 16 + (lane & 15)) * LDX + koff];
#pragma unroll
    for (int nt = 0; nt < 4; ++nt) {
      int bn = (nt * 16 + (lane & 15)) * LDW + koff;
      v16bf fq = *(const v16bf*)&wls[0][bn];
      v16bf fk = *(const v16bf*)&wls[1][bn];
      v16bf fv = *(const v16bf*)&wls[2][bn];
      accq[nt] = wmma_bf16(a, fq, accq[nt]);
      acck[nt] = wmma_bf16(a, fk, acck[nt]);
      accv[nt] = wmma_bf16(a, fv, accv[nt]);
    }
    __syncthreads();
  }

  __bf16* qh = qb + (size_t)bh * S_LEN * HS;
  __bf16* kh = kb + (size_t)bh * S_LEN * HS;
  __bf16* vh = vTb + (size_t)bh * HS * S_LEN;
  const float scale = 0.125f;  // 1/sqrt(64)
#pragma unroll
  for (int nt = 0; nt < 4; ++nt) {
    int e = nt * 16 + (lane & 15);
#pragma unroll
    for (int j = 0; j < 8; ++j) {
      int s = s0 + wave * 16 + j + 8 * (lane >> 4);
      qh[(size_t)s * HS + e] = (__bf16)(accq[nt][j] * scale);
      kh[(size_t)s * HS + e] = (__bf16)acck[nt][j];
      vh[(size_t)e * S_LEN + s] = (__bf16)accv[nt][j];
    }
  }
}

// ---------------- scores = Qs @ K^T (causal tiles only) ----------------
// 2M x 4N register blocking per wave; Q/K tiles staged by TDM.
__global__ __launch_bounds__(256) void scores_kernel(const __bf16* __restrict__ qb,
                                                     const __bf16* __restrict__ kb,
                                                     float* __restrict__ w) {
  if (blockIdx.z > blockIdx.y) return;  // strictly above diagonal: never read
  const int LDK = 64;  // unpadded: TDM packs 64-elem rows (128B, 32B-aligned)
  __shared__ __attribute__((aligned(32))) __bf16 qs[128 * LDK];
  __shared__ __attribute__((aligned(32))) __bf16 ks[128 * LDK];

  int bh = blockIdx.x;
  int s0 = blockIdx.y * 128;
  int t0 = blockIdx.z * 128;
  int tid = threadIdx.x;
  int lane = tid & 31, wave = tid >> 5;
  int wm = wave & 3, wn = wave >> 2;  // 4 x 2 wave grid -> 32 rows x 64 cols each

  const __bf16* qh = qb + (size_t)bh * S_LEN * HS;
  const __bf16* kh = kb + (size_t)bh * S_LEN * HS;

#if defined(HAVE_TDM)
  if (wave == 0) {
    tdm_load_tile_2d((unsigned)(size_t)&qs[0], &qh[(size_t)s0 * HS], 64u, 128u, 64ull);
    tdm_load_tile_2d((unsigned)(size_t)&ks[0], &kh[(size_t)t0 * HS], 64u, 128u, 64ull);
    __builtin_amdgcn_s_wait_tensorcnt(0);
  }
#else
#pragma unroll
  for (int it = 0; it < 4; ++it) {
    int i = tid + it * 256;
    int r = i >> 3, q = i & 7;
    *(float4*)&qs[r * LDK + q * 8] = *(const float4*)&qh[(size_t)(s0 + r) * HS + q * 8];
    *(float4*)&ks[r * LDK + q * 8] = *(const float4*)&kh[(size_t)(t0 + r) * HS + q * 8];
  }
#endif
  __syncthreads();

  v8f acc[2][4];
#pragma unroll
  for (int mi = 0; mi < 2; ++mi)
#pragma unroll
    for (int nt = 0; nt < 4; ++nt) acc[mi][nt] = (v8f)(0.f);

  int koff = (lane >> 4) * 16;
#pragma unroll
  for (int kc = 0; kc < 2; ++kc) {
    v16bf a0 = *(const v16bf*)&qs[(wm * 32 + (lane & 15)) * LDK + kc * 32 + koff];
    v16bf a1 = *(const v16bf*)&qs[(wm * 32 + 16 + (lane & 15)) * LDK + kc * 32 + koff];
#pragma unroll
    for (int nt = 0; nt < 4; ++nt) {
      v16bf bf = *(const v16bf*)&ks[(wn * 64 + nt * 16 + (lane & 15)) * LDK + kc * 32 + koff];
      acc[0][nt] = wmma_bf16(a0, bf, acc[0][nt]);
      acc[1][nt] = wmma_bf16(a1, bf, acc[1][nt]);
    }
  }

  float* wb = w + (size_t)bh * S_LEN * S_LEN;
#pragma unroll
  for (int mi = 0; mi < 2; ++mi)
#pragma unroll
    for (int nt = 0; nt < 4; ++nt) {
      int t = t0 + wn * 64 + nt * 16 + (lane & 15);
#pragma unroll
      for (int j = 0; j < 8; ++j) {
        int s = s0 + wm * 32 + mi * 16 + j + 8 * (lane >> 4);
        wb[(size_t)s * S_LEN + t] = acc[mi][nt][j];
      }
    }
}

// ---------------- causal softmax (in place on d_out weights) ----------------
__global__ __launch_bounds__(256) void softmax_kernel(float* __restrict__ w) {
  __shared__ float red[256];
  size_t row = blockIdx.x;
  int s = (int)(row & (S_LEN - 1));
  float* wr = w + row * (size_t)S_LEN;
  int tid = threadIdx.x;
  int nv = s + 1;

  float vals[8];
  float m = -3.4e38f;
#pragma unroll
  for (int i = 0; i < 8; ++i) {
    int t = tid + i * 256;
    vals[i] = (t < nv) ? wr[t] : -3.4e38f;
    m = fmaxf(m, vals[i]);
  }
  red[tid] = m;
  __syncthreads();
  for (int off = 128; off > 0; off >>= 1) {
    if (tid < off) red[tid] = fmaxf(red[tid], red[tid + off]);
    __syncthreads();
  }
  float M = red[0];
  __syncthreads();

  float sum = 0.f;
#pragma unroll
  for (int i = 0; i < 8; ++i) {
    int t = tid + i * 256;
    float p = (t < nv) ? __expf(vals[i] - M) : 0.f;
    vals[i] = p;
    sum += p;
  }
  red[tid] = sum;
  __syncthreads();
  for (int off = 128; off > 0; off >>= 1) {
    if (tid < off) red[tid] += red[tid + off];
    __syncthreads();
  }
  float rinv = 1.f / red[0];
#pragma unroll
  for (int i = 0; i < 8; ++i) {
    int t = tid + i * 256;
    wr[t] = vals[i] * rinv;
  }
}

// ---------------- attn = W @ V (causal k-range), bf16 out (B,S,D) ----------------
__global__ __launch_bounds__(256) void attnv_kernel(const float* __restrict__ w,
                                                    const __bf16* __restrict__ vTb,
                                                    __bf16* __restrict__ attnb) {
  const int LDK = 48;
  __shared__ __attribute__((aligned(32))) __bf16 wsh[128 * LDK];
  __shared__ __attribute__((aligned(32))) __bf16 vsh[64 * LDK];

  int bh = blockIdx.x;
  int b = bh >> 4, h = bh & 15;
  int s0 = blockIdx.y * 128;
  int tid = threadIdx.x;
  int lane = tid & 31, wave = tid >> 5;

  const float* wr = w + (size_t)bh * S_LEN * S_LEN;
  const __bf16* vh = vTb + (size_t)bh * HS * S_LEN;

  v8f acc[4];
#pragma unroll
  for (int i = 0; i < 4; ++i) acc[i] = (v8f)(0.f);

  int kend = s0 + 128;  // weights above the diagonal block are exact zeros
  for (int t0 = 0; t0 < kend; t0 += 32) {
    // stage weight tile 128 x 32 f32 -> bf16
#pragma unroll
    for (int it = 0; it < 4; ++it) {
      int i = tid + it * 256;
      int r = i >> 3, q = i & 7;
      float4 v = *(const float4*)&wr[(size_t)(s0 + r) * S_LEN + t0 + q * 4];
      __bf16 p[4] = {(__bf16)v.x, (__bf16)v.y, (__bf16)v.z, (__bf16)v.w};
      *(float2*)&wsh[r * LDK + q * 4] = *(float2*)p;
    }
    // stage V^T tile 64 x 32 bf16
    {
      int e = tid >> 2, q = tid & 3;
      *(float4*)&vsh[e * LDK + q * 8] =
          *(const float4*)&vh[(size_t)e * S_LEN + t0 + q * 8];
    }
    if (t0 + 32 < kend)
      __builtin_prefetch(&wr[(size_t)(s0 + (tid & 127)) * S_LEN + t0 + 32], 0, 1);
    __syncthreads();

    int koff = (lane >> 4) * 16;
    v16bf a = *(const v16bf*)&wsh[(wave * 16 + (lane & 15)) * LDK + koff];
#pragma unroll
    for (int nt = 0; nt < 4; ++nt) {
      v16bf bf = *(const v16bf*)&vsh[(nt * 16 + (lane & 15)) * LDK + koff];
      acc[nt] = wmma_bf16(a, bf, acc[nt]);
    }
    __syncthreads();
  }

#pragma unroll
  for (int nt = 0; nt < 4; ++nt) {
    int e = nt * 16 + (lane & 15);
#pragma unroll
    for (int j = 0; j < 8; ++j) {
      int s = s0 + wave * 16 + j + 8 * (lane >> 4);
      attnb[((size_t)b * S_LEN + s) * D_DIM + h * HS + e] = (__bf16)acc[nt][j];
    }
  }
}

// ---------------- out = attn @ Wo (2M x 4N per wave) ----------------
__global__ __launch_bounds__(256) void outproj_kernel(const __bf16* __restrict__ attnb,
                                                      const __bf16* __restrict__ woT,
                                                      float* __restrict__ out) {
  const int LDK = 48;
  __shared__ __attribute__((aligned(32))) __bf16 ash[128 * LDK];
  __shared__ __attribute__((aligned(32))) __bf16 bsh[128 * LDK];

  int r0 = blockIdx.x * 128;  // over B*S rows
  int c0 = blockIdx.y * 128;
  int tid = threadIdx.x;
  int lane = tid & 31, wave = tid >> 5;
  int wm = wave & 3, wn = wave >> 2;  // 32 rows x 64 cols per wave

  v8f acc[2][4];
#pragma unroll
  for (int mi = 0; mi < 2; ++mi)
#pragma unroll
    for (int nt = 0; nt < 4; ++nt) acc[mi][nt] = (v8f)(0.f);

  for (int k0 = 0; k0 < D_DIM; k0 += 32) {
#pragma unroll
    for (int it = 0; it < 2; ++it) {
      int i = tid + it * 256;
      int r = i >> 2, q = i & 3;
      *(float4*)&ash[r * LDK + q * 8] =
          *(const float4*)&attnb[(size_t)(r0 + r) * D_DIM + k0 + q * 8];
      *(float4*)&bsh[r * LDK + q * 8] =
          *(const float4*)&woT[(size_t)(c0 + r) * D_DIM + k0 + q * 8];
    }
    if (k0 + 32 < D_DIM)
      __builtin_prefetch(&attnb[(size_t)(r0 + (tid & 127)) * D_DIM + k0 + 32], 0, 1);
    __syncthreads();

    int koff = (lane >> 4) * 16;
    v16bf a0 = *(const v16bf*)&ash[(wm * 32 + (lane & 15)) * LDK + koff];
    v16bf a1 = *(const v16bf*)&ash[(wm * 32 + 16 + (lane & 15)) * LDK + koff];
#pragma unroll
    for (int nt = 0; nt < 4; ++nt) {
      v16bf bf = *(const v16bf*)&bsh[(wn * 64 + nt * 16 + (lane & 15)) * LDK + koff];
      acc[0][nt] = wmma_bf16(a0, bf, acc[0][nt]);
      acc[1][nt] = wmma_bf16(a1, bf, acc[1][nt]);
    }
    __syncthreads();
  }

#pragma unroll
  for (int mi = 0; mi < 2; ++mi)
#pragma unroll
    for (int nt = 0; nt < 4; ++nt) {
      int c = c0 + wn * 64 + nt * 16 + (lane & 15);
#pragma unroll
      for (int j = 0; j < 8; ++j) {
        int r = r0 + wm * 32 + mi * 16 + j + 8 * (lane >> 4);
        out[(size_t)r * D_DIM + c] = acc[mi][nt][j];
      }
    }
}

extern "C" void kernel_launch(void* const* d_in, const int* in_sizes, int n_in,
                              void* d_out, int out_size, void* d_ws, size_t ws_size,
                              hipStream_t stream) {
  const float* x  = (const float*)d_in[0];
  const float* wq = (const float*)d_in[1];
  const float* wk = (const float*)d_in[2];
  const float* wv = (const float*)d_in[3];
  const float* wo = (const float*)d_in[4];

  const size_t nX  = (size_t)B_NUM * S_LEN * D_DIM;  // 4 Mi
  const size_t nW  = (size_t)H_NUM * D_DIM * HS;     // 1 Mi per tensor
  const size_t nWo = (size_t)D_DIM * D_DIM;          // 1 Mi

  char* ws = (char*)d_ws;
  __bf16* xb   = (__bf16*)(ws);               // 8 MB
  __bf16* wqT  = (__bf16*)(ws + (8 << 20));   // 2 MB
  __bf16* wkT  = (__bf16*)(ws + (10 << 20));  // 2 MB
  __bf16* wvT  = (__bf16*)(ws + (12 << 20));  // 2 MB
  __bf16* woT  = (__bf16*)(ws + (14 << 20));  // 2 MB
  __bf16* qb   = (__bf16*)(ws + (16 << 20));  // 8 MB
  __bf16* kb   = (__bf16*)(ws + (24 << 20));  // 8 MB
  __bf16* vTb  = (__bf16*)(ws + (32 << 20));  // 8 MB
  __bf16* attb = (__bf16*)(ws + (40 << 20));  // 8 MB

  float* out = (float*)d_out;
  float* weights = out + nX;  // (B,H,S,S)

  cvt_bf16_kernel<<<(int)(nX / 256), 256, 0, stream>>>(x, xb, (int)nX);
  cvt_bf16_tr_kernel<<<(int)(nW / 256), 256, 0, stream>>>(wq, wqT, H_NUM, D_DIM, HS);
  cvt_bf16_tr_kernel<<<(int)(nW / 256), 256, 0, stream>>>(wk, wkT, H_NUM, D_DIM, HS);
  cvt_bf16_tr_kernel<<<(int)(nW / 256), 256, 0, stream>>>(wv, wvT, H_NUM, D_DIM, HS);
  cvt_bf16_tr_kernel<<<(int)(nWo / 256), 256, 0, stream>>>(wo, woT, 1, D_DIM, D_DIM);

  qkv_kernel<<<dim3(B_NUM * H_NUM, S_LEN / 128), 256, 0, stream>>>(
      xb, wqT, wkT, wvT, qb, kb, vTb);
  scores_kernel<<<dim3(B_NUM * H_NUM, S_LEN / 128, S_LEN / 128), 256, 0, stream>>>(
      qb, kb, weights);
  softmax_kernel<<<(int)((size_t)B_NUM * H_NUM * S_LEN), 256, 0, stream>>>(weights);
  attnv_kernel<<<dim3(B_NUM * H_NUM, S_LEN / 128), 256, 0, stream>>>(
      weights, vTb, attb);
  outproj_kernel<<<dim3((B_NUM * S_LEN) / 128, D_DIM / 128), 256, 0, stream>>>(
      attb, woT, out);

  (void)in_sizes; (void)n_in; (void)out_size; (void)ws_size;
}
